// NeuralTensorNetwork_41386304864278
// MI455X (gfx1250) — matrix operand order repaired and checked by reading, not compiled, not run on hardware
//
#include <hip/hip_runtime.h>

// Workspace layout (requires ~42.3 MB):
//   Tb   bf16[256*256*256]  @ 0         (33554432 B)
//   e1b  bf16[4096*256]     @ 33554432  ( 2097152 B)
//   e2b  bf16[4096*256]     @ 35651584  ( 2097152 B)
//   W1b  bf16[256*512]      @ 37748736  (  262144 B)
//   r    f32 [256*4096]     @ 38010880  ( 4194304 B)   (stored [k, b])

typedef __attribute__((ext_vector_type(16))) __bf16 v16bf;
typedef __attribute__((ext_vector_type(8)))  float  v8f;

union Frag {
  uint4 q[2];
  v16bf v;
};

__device__ __forceinline__ unsigned f32_to_bf16u(float f) {
  __bf16 h = (__bf16)f;   // RNE convert
  return (unsigned)__builtin_bit_cast(unsigned short, h);
}

// ---------- f32 -> bf16 pack, 8 elements per thread ----------
__global__ __launch_bounds__(256) void k_cvt(const float* __restrict__ src,
                                             unsigned short* __restrict__ dst,
                                             int n8) {
  int idx = blockIdx.x * 256 + threadIdx.x;
  if (idx >= n8) return;
  const float4* p = (const float4*)src + ((size_t)idx << 1);
  float4 a = p[0], b = p[1];
  uint4 o;
  o.x = f32_to_bf16u(a.x) | (f32_to_bf16u(a.y) << 16);
  o.y = f32_to_bf16u(a.z) | (f32_to_bf16u(a.w) << 16);
  o.z = f32_to_bf16u(b.x) | (f32_to_bf16u(b.y) << 16);
  o.w = f32_to_bf16u(b.z) | (f32_to_bf16u(b.w) << 16);
  ((uint4*)dst)[idx] = o;
}

// ---------- fused bilinear + linear GEMM, output r[k, b] ----------
// D = A x B per wave:  A = 16(k) x 32(red) from Tb / W1b,
//                      B = 32(red) x 16(b) synthesized: e1[b,i]*e2[b,j] (or raw e)
// Wave covers MT=2 k-subtiles x NT=4 batch-tiles (32k x 64b of C).
// Block = 128 threads (4 waves) -> 128 k rows; grid = (64 b-groups, 2 k-halves).
__global__ __launch_bounds__(128) void k_ntn_gemm(
    const unsigned short* __restrict__ Tb,
    const unsigned short* __restrict__ e1b,
    const unsigned short* __restrict__ e2b,
    const unsigned short* __restrict__ W1b,
    const float* __restrict__ w1bias,
    float* __restrict__ r)
{
  const int lane = threadIdx.x & 31;
  const int wave = threadIdx.x >> 5;
  const int ln   = lane & 15;
  const bool hi  = lane >= 16;
  const int koff = hi ? 8 : 0;    // B-frag: runs K=[koff,koff+8) and K=[koff+16,koff+24)
  const int aoff = hi ? 8  : 0;   // A-frag: runs K=[aoff,aoff+8) and K=[aoff+16,aoff+24)
  const int b0   = blockIdx.x * 64;
  const int kb   = blockIdx.y * 128 + wave * 32;

  const __bf16* __restrict__ e1h = (const __bf16*)e1b;

  v8f acc[2][4];
#pragma unroll
  for (int s = 0; s < 2; ++s)
#pragma unroll
    for (int t = 0; t < 4; ++t)
      acc[s][t] = (v8f){};

  const size_t rowT = (size_t)(kb + ln) * 256;

  // ===== bilinear: reduction over (i in [0,256), j-chunks of 32) =====
  for (int jc = 0; jc < 8; ++jc) {
    const int j0 = jc * 32;
    Frag e2c[4];                       // lane's two 8-value e2 runs (K=[koff,koff+8), [koff+16,koff+24)) per batch-tile
#pragma unroll
    for (int t = 0; t < 4; ++t) {
      const unsigned short* p = e2b + (size_t)(b0 + t * 16 + ln) * 256 + j0 + koff;
      e2c[t].q[0] = *(const uint4*)p;
      e2c[t].q[1] = *(const uint4*)(p + 16);
    }
#pragma unroll 2
    for (int i = 0; i < 256; ++i) {
      // A fragments (T rows for this wave's two k-subtiles)
      const unsigned short* pa = Tb + ((size_t)i << 16) + rowT + (j0 + aoff);
      Frag a0, a1;
      a0.q[0] = *(const uint4*)pa;
      a0.q[1] = *(const uint4*)(pa + 16);
      a1.q[0] = *(const uint4*)(pa + 16 * 256);
      a1.q[1] = *(const uint4*)(pa + 16 * 256 + 16);

      // Per-batch-tile e1 scalars (bf16 loads, broadcast via op_sel in pk_mul)
      __bf16 sb[4];
#pragma unroll
      for (int t = 0; t < 4; ++t)
        sb[t] = e1h[(size_t)(b0 + t * 16 + ln) * 256 + i];

      // Build all four rank-1 B fragments first (keeps WAR distance to WMMAs)
      Frag bfr[4];
#pragma unroll
      for (int t = 0; t < 4; ++t)
        bfr[t].v = e2c[t].v * sb[t];   // v_pk_mul_bf16 x8, op_sel broadcast

      // Then the 8 WMMAs back-to-back
#pragma unroll
      for (int t = 0; t < 4; ++t) {
        acc[0][t] = __builtin_amdgcn_wmma_f32_16x16x32_bf16(
            false, a0.v, false, bfr[t].v, (short)0, acc[0][t], false, false);
        acc[1][t] = __builtin_amdgcn_wmma_f32_16x16x32_bf16(
            false, a1.v, false, bfr[t].v, (short)0, acc[1][t], false, false);
      }
    }
  }

  // ===== linear: reduction over m in [0,512), chunks of 32 =====
  const size_t rowW = (size_t)(kb + ln) * 512;
#pragma unroll
  for (int mc = 0; mc < 16; ++mc) {
    const int m0 = mc * 32;
    const unsigned short* pa = W1b + rowW + (m0 + aoff);
    Frag a0, a1;
    a0.q[0] = *(const uint4*)pa;
    a0.q[1] = *(const uint4*)(pa + 16);
    a1.q[0] = *(const uint4*)(pa + 16 * 512);
    a1.q[1] = *(const uint4*)(pa + 16 * 512 + 16);
    const unsigned short* src = (mc < 8) ? e1b : e2b;
    const int moff = (m0 & 255) + koff;
#pragma unroll
    for (int t = 0; t < 4; ++t) {
      const unsigned short* pb = src + (size_t)(b0 + t * 16 + ln) * 256 + moff;
      Frag bf;
      bf.q[0] = *(const uint4*)pb;
      bf.q[1] = *(const uint4*)(pb + 16);
      acc[0][t] = __builtin_amdgcn_wmma_f32_16x16x32_bf16(
          false, a0.v, false, bf.v, (short)0, acc[0][t], false, false);
      acc[1][t] = __builtin_amdgcn_wmma_f32_16x16x32_bf16(
          false, a1.v, false, bf.v, (short)0, acc[1][t], false, false);
    }
  }

  // ===== bias + store r[k, b] (C layout: VGPR v -> M = v or v+8) =====
#pragma unroll
  for (int s = 0; s < 2; ++s) {
#pragma unroll
    for (int v = 0; v < 8; ++v) {
      const int k = kb + s * 16 + (hi ? v + 8 : v);
      const float bias = w1bias[k];
      float* rp = r + (size_t)k * 4096 + b0 + ln;
#pragma unroll
      for (int t = 0; t < 4; ++t)
        rp[t * 16] = acc[s][t][v] + bias;
    }
  }
}

// ---------- BatchNorm (training stats, biased var) + tanh, transpose to [b, k] ----------
__global__ __launch_bounds__(256) void k_bn_tanh(const float* __restrict__ r,
                                                 const float* __restrict__ gamma,
                                                 const float* __restrict__ beta,
                                                 float* __restrict__ out) {
  __shared__ float ssum[256];
  __shared__ float ssq[256];
  const int k = blockIdx.x;
  const int t = threadIdx.x;
  const float* row = r + (size_t)k * 4096;
  float s = 0.f, q = 0.f;
#pragma unroll
  for (int j = 0; j < 16; ++j) {
    float v = row[t + j * 256];
    s += v;
    q += v * v;
  }
  ssum[t] = s;
  ssq[t]  = q;
  __syncthreads();
  for (int off = 128; off > 0; off >>= 1) {
    if (t < off) { ssum[t] += ssum[t + off]; ssq[t] += ssq[t + off]; }
    __syncthreads();
  }
  const float mean  = ssum[0] * (1.0f / 4096.0f);
  const float var   = ssq[0] * (1.0f / 4096.0f) - mean * mean;
  const float scale = gamma[k] * rsqrtf(var + 1e-5f);
  const float shift = beta[k] - mean * scale;
#pragma unroll
  for (int j = 0; j < 16; ++j) {
    const int b = t + j * 256;
    out[(size_t)b * 256 + k] = tanhf(row[b] * scale + shift);
  }
}

extern "C" void kernel_launch(void* const* d_in, const int* in_sizes, int n_in,
                              void* d_out, int out_size, void* d_ws, size_t ws_size,
                              hipStream_t stream) {
  (void)in_sizes; (void)n_in; (void)out_size; (void)ws_size;
  const float* emb1 = (const float*)d_in[0];
  const float* emb2 = (const float*)d_in[1];
  const float* T1   = (const float*)d_in[2];
  const float* W1w  = (const float*)d_in[3];
  const float* W1bv = (const float*)d_in[4];
  const float* gam  = (const float*)d_in[5];
  const float* bet  = (const float*)d_in[6];

  char* ws = (char*)d_ws;
  unsigned short* Tb   = (unsigned short*)(ws);
  unsigned short* e1b  = (unsigned short*)(ws + 33554432);
  unsigned short* e2b  = (unsigned short*)(ws + 35651584);
  unsigned short* W1b  = (unsigned short*)(ws + 37748736);
  float*          rbuf = (float*)(ws + 38010880);

  // bf16 conversions (streamed once; T1 then lives in L2 at 32 MB)
  k_cvt<<<8192, 256, 0, stream>>>(T1,   Tb,  2097152);
  k_cvt<<<512,  256, 0, stream>>>(emb1, e1b, 131072);
  k_cvt<<<512,  256, 0, stream>>>(emb2, e2b, 131072);
  k_cvt<<<64,   256, 0, stream>>>(W1w,  W1b, 16384);

  // fused bilinear + linear (+bias) -> r[k, b]
  k_ntn_gemm<<<dim3(64, 2), 128, 0, stream>>>(Tb, e1b, e2b, W1b, W1bv, rbuf);

  // batchnorm + tanh -> out[b, k]
  k_bn_tanh<<<256, 256, 0, stream>>>(rbuf, gam, bet, (float*)d_out);
}
